// GGNNNet_36318243455514
// MI455X (gfx1250) — matrix operand order repaired and checked by reading, not compiled, not run on hardware
//
#include <hip/hip_runtime.h>
#include <hip/hip_bf16.h>

// ---------------------------------------------------------------------------
// GGNN on gfx1250 (MI455X).  Dense ops via V_WMMA_F32_16X16X32_F16 (wave32):
//  - pre-converted f16 weights (B-fragment = 2x global_load_b128, zero cvt)
//  - TWO 16-row tiles per wave -> each B-fragment feeds 2 WMMAs (halves the
//    L2 weight traffic, the dominant dense-phase byte stream)
//  - biases folded into WMMA C seeds; zero-seeded accs use inline C=0
// Irregular ops: f16 gathers + global_atomic_add_f32 scatters (HBM-bound).
// ---------------------------------------------------------------------------

#define GG_N      50000
#define GG_E      600000
#define GG_G      64
#define GG_DIN    128
#define GG_DHID   256
#define GG_DOUT   128
#define GG_NDIFF  1000
#define GG_TILES  (GG_N / 16)          // 3125
#define GG_GRID2  ((GG_TILES + 1) / 2) // 1563 (two tiles per wave)

typedef __attribute__((ext_vector_type(16))) _Float16 v16h;
typedef __attribute__((ext_vector_type(8)))  float    v8f;

#define LDX 136   // 128 + 8 halves pad (A tiles, K=128)
#define LDH 264   // 256 + 8 halves pad (hidden tile, K=256)

__device__ inline v8f wmma_f16(v16h a, v16h b, v8f c) {
  // D = A(16x32 f16) * B(32x16 f16) + C(16x16 f32)
  return __builtin_amdgcn_wmma_f32_16x16x32_f16(
      /*neg_a=*/false, a, /*neg_b=*/false, b,
      /*c_mod=*/(short)0, c, /*reuse_a=*/false, /*reuse_b=*/false);
}

// A-fragment (16x32, f16) from a row-major f16 LDS tile, row stride `ld`
// halves, K-column offset already folded into `row0`.
// Contiguous K order (matches frag_b_h16): lane m=lane&15; halves 0..15 = K (lane>>4)*16 ..+15.
__device__ inline v16h frag_a_lds(const _Float16* row0, int ld) {
  const int lane = threadIdx.x & 31;
  const _Float16* r = row0 + (lane & 15) * ld;
  const int hb = (lane >> 4) * 16;
  v16h a;
#pragma unroll
  for (int i = 0; i < 8; ++i) a[i] = r[hb + i];
#pragma unroll
  for (int i = 0; i < 8; ++i) a[8 + i] = r[hb + 8 + i];
  return a;
}

// B-fragment (32x16, f16) where B = W^T for a row-major *f16* weight
// Wh[Nout][K]: column n of B == row n of Wh -> 16 contiguous halves per lane
// (32 bytes, 32B-aligned) -> two global_load_b128, zero conversion VALU.
// Contiguous K order (matches frag_a_lds): lane n=lane&15; halves 0..15 = K (lane>>4)*16 ..+15.
__device__ inline v16h frag_b_h16(const _Float16* __restrict__ Wh,
                                  int n0, int ldw, int k0) {
  const int lane = threadIdx.x & 31;
  const _Float16* r =
      Wh + (size_t)(n0 + (lane & 15)) * ldw + k0 + (lane >> 4) * 16;
  return *(const v16h*)r;
}

// Seed the WMMA accumulator with a per-column bias: in the f32 C/D layout the
// column index (lane&15) is fixed per lane, so all 8 row-VGPRs share `bias`.
__device__ inline v8f acc_bias(float bias) {
  v8f c;
#pragma unroll
  for (int r = 0; r < 8; ++r) c[r] = bias;
  return c;
}

__device__ inline float softplus_f(float x) {
  return x > 20.0f ? x : log1pf(__expf(x));
}
__device__ inline float sigmoid_f(float x) {
  return 1.0f / (1.0f + __expf(-x));
}
__device__ inline void atomic_add_f32(float* p, float v) {
  __hip_atomic_fetch_add(p, v, __ATOMIC_RELAXED, __HIP_MEMORY_SCOPE_AGENT);
}

// ---------------------------------------------------------------------------
// Weight f32 -> f16 pre-conversion (once per launch).  n4 = count / 4.
// ---------------------------------------------------------------------------
__global__ void ggnn_cvt_f16_kernel(const float* __restrict__ src,
                                    _Float16* __restrict__ dst, int n4) {
  int i = blockIdx.x * blockDim.x + threadIdx.x;
  if (i >= n4) return;
  float4 v = ((const float4*)src)[i];
  _Float16 h4[4] = {(_Float16)v.x, (_Float16)v.y, (_Float16)v.z, (_Float16)v.w};
  *(uint2*)(dst + (size_t)i * 4) = *(const uint2*)h4;
}

// ---------------------------------------------------------------------------
// Degree / alpha prep
// ---------------------------------------------------------------------------
__global__ void ggnn_deg_kernel(const int* __restrict__ ei, float* __restrict__ deg) {
  int e = blockIdx.x * blockDim.x + threadIdx.x;
  if (e < GG_E) atomic_add_f32(&deg[ei[GG_E + e]], 1.0f);  // dst = ei[1][e]
}

__global__ void ggnn_invdeg_kernel(float* __restrict__ deg) {
  int i = blockIdx.x * blockDim.x + threadIdx.x;
  if (i < GG_N) deg[i] = 1.0f / fmaxf(deg[i], 1.0f);
}

__global__ void ggnn_alpha_init_kernel(float* __restrict__ alpha) {
  int i = blockIdx.x * blockDim.x + threadIdx.x;
  if (i < GG_N) alpha[i] = 1.0f;
}

__global__ void ggnn_alpha_set_kernel(float* __restrict__ alpha,
                                      const int* __restrict__ diff_idx,
                                      const float* __restrict__ w_imp) {
  int i = blockIdx.x * blockDim.x + threadIdx.x;
  if (i < GG_NDIFF) alpha[diff_idx[i]] = 1.0f + softplus_f(*w_imp);
}

// ---------------------------------------------------------------------------
// Input MLP: h = relu(x @ W1^T + b1) @ W2^T + b2
// One wave per TWO 16-node tiles: each B-fragment is used for 2 WMMAs.
// ---------------------------------------------------------------------------
__global__ __launch_bounds__(32) void ggnn_mlp_kernel(
    const float* __restrict__ x,
    const _Float16* __restrict__ w1h, const float* __restrict__ b1,
    const _Float16* __restrict__ w2h, const float* __restrict__ b2,
    float* __restrict__ h) {
  __shared__ _Float16 xs[2 * 16 * LDX];   // 2 x (16 x 128) input tiles (f16)
  __shared__ _Float16 hs[2 * 16 * LDH];   // 2 x (16 x 256) hidden tiles
  const int tile0 = blockIdx.x * 2;
  const int t     = threadIdx.x;
  const int lane  = t & 31;

  // stage both x tiles -> f16 LDS (row clamped; tail tile guarded at store)
  for (int idx = t; idx < 2 * 16 * 128; idx += 32) {
    int tt = idx >> 11, rem = idx & 2047;
    int r = rem >> 7, c = rem & 127;
    int row = (tile0 + tt) * 16 + r;
    row = row < GG_N ? row : GG_N - 1;
    xs[tt * 16 * LDX + r * LDX + c] = (_Float16)x[(size_t)row * GG_DIN + c];
  }
  __syncthreads();

  // layer 1: [16x128] @ [128x256] + b1 -> ReLU -> hs
  for (int jt = 0; jt < GG_DHID / 16; ++jt) {
    const int n = jt * 16 + (lane & 15);
    v8f acc[2];
    acc[0] = acc_bias(b1[n]);
    acc[1] = acc[0];
#pragma unroll
    for (int k0 = 0; k0 < GG_DIN; k0 += 32) {
      v16h b = frag_b_h16(w1h, jt * 16, GG_DIN, k0);
#pragma unroll
      for (int tt = 0; tt < 2; ++tt)
        acc[tt] = wmma_f16(frag_a_lds(xs + tt * 16 * LDX + k0, LDX), b, acc[tt]);
    }
    const int rbase = (lane >> 4) * 8;
#pragma unroll
    for (int tt = 0; tt < 2; ++tt)
#pragma unroll
      for (int r = 0; r < 8; ++r) {
        float v = acc[tt][r];
        hs[tt * 16 * LDH + (rbase + r) * LDH + n] =
            (_Float16)(v > 0.0f ? v : 0.0f);
      }
  }
  __syncthreads();

  // layer 2: [16x256] @ [256x128] + b2 -> h (f32, global)
  for (int jt = 0; jt < GG_DOUT / 16; ++jt) {
    const int n = jt * 16 + (lane & 15);
    v8f acc[2];
    acc[0] = acc_bias(b2[n]);
    acc[1] = acc[0];
#pragma unroll
    for (int k0 = 0; k0 < GG_DHID; k0 += 32) {
      v16h b = frag_b_h16(w2h, jt * 16, GG_DHID, k0);
#pragma unroll
      for (int tt = 0; tt < 2; ++tt)
        acc[tt] = wmma_f16(frag_a_lds(hs + tt * 16 * LDH + k0, LDH), b, acc[tt]);
    }
    const int rbase = (lane >> 4) * 8;
#pragma unroll
    for (int tt = 0; tt < 2; ++tt)
#pragma unroll
      for (int r = 0; r < 8; ++r) {
        int row = (tile0 + tt) * 16 + rbase + r;
        if (row < GG_N) h[(size_t)row * GG_DOUT + n] = acc[tt][r];
      }
  }
}

// ---------------------------------------------------------------------------
// LayerNorm (no affine): h <- (h-mu)*rsqrt(var+eps); also emit f16 copy.
// One wave per row; wave32 shuffle reductions.
// ---------------------------------------------------------------------------
__global__ __launch_bounds__(256) void ggnn_layernorm_kernel(
    float* __restrict__ h, _Float16* __restrict__ hn16) {
  int wave = (blockIdx.x * blockDim.x + threadIdx.x) >> 5;
  int lane = threadIdx.x & 31;
  if (wave >= GG_N) return;
  float* row = h + (size_t)wave * GG_DOUT;
  float v[4];
  float s = 0.0f;
#pragma unroll
  for (int i = 0; i < 4; ++i) { v[i] = row[lane * 4 + i]; s += v[i]; }
#pragma unroll
  for (int o = 16; o > 0; o >>= 1) s += __shfl_xor(s, o, 32);
  const float mu = s * (1.0f / 128.0f);
  float q = 0.0f;
#pragma unroll
  for (int i = 0; i < 4; ++i) { v[i] -= mu; q += v[i] * v[i]; }
#pragma unroll
  for (int o = 16; o > 0; o >>= 1) q += __shfl_xor(q, o, 32);
  const float inv = rsqrtf(q * (1.0f / 128.0f) + 1e-5f);
  _Float16* r16 = hn16 + (size_t)wave * GG_DOUT;
#pragma unroll
  for (int i = 0; i < 4; ++i) {
    float nv = v[i] * inv;
    row[lane * 4 + i] = nv;
    r16[lane * 4 + i] = (_Float16)nv;
  }
}

// ---------------------------------------------------------------------------
// Edge scatter: m[dst] += softplus(edge_w[type]) * hn[src]
// One wave per edge; f16 gather (b64), f32 HW atomics for accumulation.
// ---------------------------------------------------------------------------
__global__ __launch_bounds__(256) void ggnn_scatter_kernel(
    const int* __restrict__ ei, const int* __restrict__ etype,
    const float* __restrict__ edge_w, const _Float16* __restrict__ hn16,
    float* __restrict__ m) {
  int wave = (blockIdx.x * blockDim.x + threadIdx.x) >> 5;
  int lane = threadIdx.x & 31;
  if (wave >= GG_E) return;
  const int src = ei[wave];
  const int dst = ei[GG_E + wave];
  const float w = softplus_f(edge_w[etype[wave]]);
  const _Float16* hr = hn16 + (size_t)src * GG_DOUT + lane * 4;
  float* mr = m + (size_t)dst * GG_DOUT + lane * 4;
  uint2 packed = *(const uint2*)hr;       // 4 halves, one b64 load
  const _Float16* hv = (const _Float16*)&packed;
#pragma unroll
  for (int i = 0; i < 4; ++i) atomic_add_f32(&mr[i], w * (float)hv[i]);
}

// ---------------------------------------------------------------------------
// GRU cell, TWO 16-node tiles per wave (B-fragment reuse across tiles):
//   gi = (m*invdeg) @ Wih^T + bih ; gh = hn @ Whh^T + bhh
//   r=sig(ir+hr) z=sig(iz+hz) n=tanh(in+r*h_n) ; h = (1-z)*n + z*h
// bhh_r/bhh_z folded into ir/iz seeds; hr/hz zero-seeded (inline C=0).
// ---------------------------------------------------------------------------
__global__ __launch_bounds__(32) void ggnn_gru_kernel(
    const float* __restrict__ m, const float* __restrict__ invdeg,
    const _Float16* __restrict__ hn16, float* __restrict__ h,
    const _Float16* __restrict__ wihh, const _Float16* __restrict__ whhh,
    const float* __restrict__ bih, const float* __restrict__ bhh) {
  __shared__ _Float16 ms[2 * 16 * LDX];
  __shared__ _Float16 hsh[2 * 16 * LDX];
  const int tile0 = blockIdx.x * 2;
  const int t     = threadIdx.x;
  const int lane  = t & 31;

  // stage m (scaled by invdeg) and hn (f16) for both tiles
  for (int idx = t; idx < 2 * 16 * 32; idx += 32) {
    int tt = idx >> 9, rem = idx & 511;
    int r = rem >> 5, c4 = (rem & 31) * 4;
    int row = (tile0 + tt) * 16 + r;
    row = row < GG_N ? row : GG_N - 1;
    float sc = invdeg[row];
    float4 mv = *(const float4*)(m + (size_t)row * GG_DOUT + c4);
    _Float16* d = ms + tt * 16 * LDX + r * LDX + c4;
    d[0] = (_Float16)(mv.x * sc); d[1] = (_Float16)(mv.y * sc);
    d[2] = (_Float16)(mv.z * sc); d[3] = (_Float16)(mv.w * sc);
    *(uint2*)(hsh + tt * 16 * LDX + r * LDX + c4) =
        *(const uint2*)(hn16 + (size_t)row * GG_DOUT + c4);
  }
  __syncthreads();

  for (int jt = 0; jt < GG_DOUT / 16; ++jt) {
    const int n = jt * 16 + (lane & 15);
    const float biR = bih[n] + bhh[n];
    const float biZ = bih[128 + n] + bhh[128 + n];
    const float biG = bih[256 + n];
    const float bhG = bhh[256 + n];
    v8f ir[2], iz[2], ig[2], hr[2], hz[2], hg[2];
#pragma unroll
    for (int tt = 0; tt < 2; ++tt) {
      ir[tt] = acc_bias(biR);
      iz[tt] = acc_bias(biZ);
      ig[tt] = acc_bias(biG);
      hr[tt] = (v8f){};          // inline C=0
      hz[tt] = (v8f){};          // inline C=0
      hg[tt] = acc_bias(bhG);    // bhh_n must stay inside r * h_n
    }
#pragma unroll
    for (int k0 = 0; k0 < GG_DOUT; k0 += 32) {
      v16h am0 = frag_a_lds(ms + k0, LDX);
      v16h am1 = frag_a_lds(ms + 16 * LDX + k0, LDX);
      v16h ah0 = frag_a_lds(hsh + k0, LDX);
      v16h ah1 = frag_a_lds(hsh + 16 * LDX + k0, LDX);
      v16h b;
      b = frag_b_h16(wihh, jt * 16, GG_DOUT, k0);
      ir[0] = wmma_f16(am0, b, ir[0]); ir[1] = wmma_f16(am1, b, ir[1]);
      b = frag_b_h16(wihh, 128 + jt * 16, GG_DOUT, k0);
      iz[0] = wmma_f16(am0, b, iz[0]); iz[1] = wmma_f16(am1, b, iz[1]);
      b = frag_b_h16(wihh, 256 + jt * 16, GG_DOUT, k0);
      ig[0] = wmma_f16(am0, b, ig[0]); ig[1] = wmma_f16(am1, b, ig[1]);
      b = frag_b_h16(whhh, jt * 16, GG_DOUT, k0);
      hr[0] = wmma_f16(ah0, b, hr[0]); hr[1] = wmma_f16(ah1, b, hr[1]);
      b = frag_b_h16(whhh, 128 + jt * 16, GG_DOUT, k0);
      hz[0] = wmma_f16(ah0, b, hz[0]); hz[1] = wmma_f16(ah1, b, hz[1]);
      b = frag_b_h16(whhh, 256 + jt * 16, GG_DOUT, k0);
      hg[0] = wmma_f16(ah0, b, hg[0]); hg[1] = wmma_f16(ah1, b, hg[1]);
    }
    const int rbase = (lane >> 4) * 8;
#pragma unroll
    for (int tt = 0; tt < 2; ++tt)
#pragma unroll
      for (int r = 0; r < 8; ++r) {
        int row = (tile0 + tt) * 16 + rbase + r;
        if (row < GG_N) {
          const size_t ofs = (size_t)row * GG_DOUT + n;
          const float hv = h[ofs];  // normalized h (GRU hidden state input)
          const float rg = sigmoid_f(ir[tt][r] + hr[tt][r]);
          const float zg = sigmoid_f(iz[tt][r] + hz[tt][r]);
          const float ng = tanhf(ig[tt][r] + rg * hg[tt][r]);
          h[ofs] = (1.0f - zg) * ng + zg * hv;
        }
      }
  }
}

// ---------------------------------------------------------------------------
// Importance-weighted pooling: xsum[batch[n]] += alpha[n]*h[n]; asum += alpha
// ---------------------------------------------------------------------------
__global__ __launch_bounds__(256) void ggnn_pool_kernel(
    const float* __restrict__ h, const float* __restrict__ alpha,
    const int* __restrict__ batch, float* __restrict__ xsum,
    float* __restrict__ asum) {
  int wave = (blockIdx.x * blockDim.x + threadIdx.x) >> 5;
  int lane = threadIdx.x & 31;
  if (wave >= GG_N) return;
  const float a = alpha[wave];
  const int g = batch[wave];
  const float* hr = h + (size_t)wave * GG_DOUT + lane * 4;
  float* xr = xsum + (size_t)g * GG_DOUT + lane * 4;
#pragma unroll
  for (int i = 0; i < 4; ++i) atomic_add_f32(&xr[i], a * hr[i]);
  if (lane == 0) atomic_add_f32(&asum[g], a);
}

// ---------------------------------------------------------------------------
// Classifier head (tiny): xg = xsum/asum; out = fc3(relu(fc2(relu(fc1(xg)))))
// Single block, one thread per graph.
// ---------------------------------------------------------------------------
__global__ __launch_bounds__(64) void ggnn_head_kernel(
    const float* __restrict__ xsum, const float* __restrict__ asum,
    const float* __restrict__ f1w, const float* __restrict__ f1b,
    const float* __restrict__ f2w, const float* __restrict__ f2b,
    const float* __restrict__ f3w, const float* __restrict__ f3b,
    float* __restrict__ out) {
  __shared__ float xg[GG_G][GG_DOUT];
  __shared__ float o1[GG_G][64];
  const int t = threadIdx.x;
  for (int idx = t; idx < GG_G * GG_DOUT; idx += 64) {
    int g = idx >> 7, c = idx & 127;
    xg[g][c] = xsum[g * GG_DOUT + c] / asum[g];
  }
  __syncthreads();
  const int g = t;
  for (int j = 0; j < 64; ++j) {
    float s = f1b[j];
#pragma unroll 4
    for (int k = 0; k < GG_DOUT; ++k) s += xg[g][k] * f1w[j * GG_DOUT + k];
    o1[g][j] = s > 0.0f ? s : 0.0f;
  }
  __syncthreads();
  float o2[32];
  for (int j = 0; j < 32; ++j) {
    float s = f2b[j];
#pragma unroll 4
    for (int k = 0; k < 64; ++k) s += o1[g][k] * f2w[j * 64 + k];
    o2[j] = s > 0.0f ? s : 0.0f;
  }
#pragma unroll
  for (int j = 0; j < 2; ++j) {
    float s = f3b[j];
#pragma unroll
    for (int k = 0; k < 32; ++k) s += o2[k] * f3w[j * 32 + k];
    out[g * 2 + j] = s;
  }
}

// ---------------------------------------------------------------------------
// Host launcher
// ---------------------------------------------------------------------------
extern "C" void kernel_launch(void* const* d_in, const int* in_sizes, int n_in,
                              void* d_out, int out_size, void* d_ws, size_t ws_size,
                              hipStream_t stream) {
  const float* x       = (const float*)d_in[0];
  const float* lin1_w  = (const float*)d_in[1];
  const float* lin1_b  = (const float*)d_in[2];
  const float* lin2_w  = (const float*)d_in[3];
  const float* lin2_b  = (const float*)d_in[4];
  const float* gru_wih = (const float*)d_in[5];
  const float* gru_whh = (const float*)d_in[6];
  const float* gru_bih = (const float*)d_in[7];
  const float* gru_bhh = (const float*)d_in[8];
  const float* edge_w  = (const float*)d_in[9];
  const float* w_imp   = (const float*)d_in[10];
  const float* fc1_w   = (const float*)d_in[11];
  const float* fc1_b   = (const float*)d_in[12];
  const float* fc2_w   = (const float*)d_in[13];
  const float* fc2_b   = (const float*)d_in[14];
  const float* fc3_w   = (const float*)d_in[15];
  const float* fc3_b   = (const float*)d_in[16];
  const int*   ei      = (const int*)d_in[17];   // [2, E]
  const int*   etype   = (const int*)d_in[18];   // [E]
  const int*   batch   = (const int*)d_in[19];   // [N]
  const int*   diffidx = (const int*)d_in[20];   // [N_DIFF]
  float* out = (float*)d_out;                    // [G, 2]

  // Workspace carve-up (256B aligned)
  char* ws = (char*)d_ws;
  size_t off = 0;
  auto carve = [&](size_t bytes) {
    void* p = ws + off;
    off += (bytes + 255) & ~(size_t)255;
    return p;
  };
  float*     h      = (float*)carve((size_t)GG_N * GG_DOUT * sizeof(float));
  _Float16*  hn16   = (_Float16*)carve((size_t)GG_N * GG_DOUT * sizeof(_Float16));
  float*     msgbuf = (float*)carve((size_t)GG_N * GG_DOUT * sizeof(float));
  float*     invdeg = (float*)carve((size_t)GG_N * sizeof(float));
  float*     alpha  = (float*)carve((size_t)GG_N * sizeof(float));
  float*     xsum   = (float*)carve((size_t)GG_G * GG_DOUT * sizeof(float));
  float*     asum   = (float*)carve((size_t)GG_G * sizeof(float));
  _Float16*  w1h    = (_Float16*)carve((size_t)GG_DHID * GG_DIN * sizeof(_Float16));
  _Float16*  w2h    = (_Float16*)carve((size_t)GG_DOUT * GG_DHID * sizeof(_Float16));
  _Float16*  wihh   = (_Float16*)carve((size_t)3 * GG_DOUT * GG_DOUT * sizeof(_Float16));
  _Float16*  whhh   = (_Float16*)carve((size_t)3 * GG_DOUT * GG_DOUT * sizeof(_Float16));
  (void)ws_size; (void)n_in; (void)in_sizes; (void)out_size;

  // --- weight pre-conversion to f16 ---------------------------------------
  {
    const int n1 = GG_DHID * GG_DIN / 4;        // 8192
    const int n2 = GG_DOUT * GG_DHID / 4;       // 8192
    const int n3 = 3 * GG_DOUT * GG_DOUT / 4;   // 12288
    ggnn_cvt_f16_kernel<<<(n1 + 255) / 256, 256, 0, stream>>>(lin1_w, w1h, n1);
    ggnn_cvt_f16_kernel<<<(n2 + 255) / 256, 256, 0, stream>>>(lin2_w, w2h, n2);
    ggnn_cvt_f16_kernel<<<(n3 + 255) / 256, 256, 0, stream>>>(gru_wih, wihh, n3);
    ggnn_cvt_f16_kernel<<<(n3 + 255) / 256, 256, 0, stream>>>(gru_whh, whhh, n3);
  }

  // --- degree / alpha prep -------------------------------------------------
  hipMemsetAsync(invdeg, 0, (size_t)GG_N * sizeof(float), stream);
  ggnn_deg_kernel<<<(GG_E + 255) / 256, 256, 0, stream>>>(ei, invdeg);
  ggnn_invdeg_kernel<<<(GG_N + 255) / 256, 256, 0, stream>>>(invdeg);
  ggnn_alpha_init_kernel<<<(GG_N + 255) / 256, 256, 0, stream>>>(alpha);
  ggnn_alpha_set_kernel<<<(GG_NDIFF + 255) / 256, 256, 0, stream>>>(alpha, diffidx, w_imp);

  // --- input MLP (WMMA, 2 tiles/wave) --------------------------------------
  ggnn_mlp_kernel<<<GG_GRID2, 32, 0, stream>>>(x, w1h, lin1_b, w2h, lin2_b, h);

  // --- 3 GNN layers --------------------------------------------------------
  for (int layer = 0; layer < 3; ++layer) {
    ggnn_layernorm_kernel<<<(GG_N * 32 + 255) / 256, 256, 0, stream>>>(h, hn16);
    hipMemsetAsync(msgbuf, 0, (size_t)GG_N * GG_DOUT * sizeof(float), stream);
    ggnn_scatter_kernel<<<(GG_E * 32 + 255) / 256, 256, 0, stream>>>(
        ei, etype, edge_w, hn16, msgbuf);
    ggnn_gru_kernel<<<GG_GRID2, 32, 0, stream>>>(
        msgbuf, invdeg, hn16, h, wihh, whhh, gru_bih, gru_bhh);
  }

  // --- pooling + head ------------------------------------------------------
  hipMemsetAsync(xsum, 0, (size_t)GG_G * GG_DOUT * sizeof(float), stream);
  hipMemsetAsync(asum, 0, (size_t)GG_G * sizeof(float), stream);
  ggnn_pool_kernel<<<(GG_N * 32 + 255) / 256, 256, 0, stream>>>(h, alpha, batch, xsum, asum);
  ggnn_head_kernel<<<1, 64, 0, stream>>>(xsum, asum, fc1_w, fc1_b, fc2_w, fc2_b,
                                         fc3_w, fc3_b, out);
}
